// MultiheadSDE_layer_70720931496058
// MI455X (gfx1250) — matrix operand order repaired and checked by reading, not compiled, not run on hardware
//
#include <hip/hip_runtime.h>
#include <hip/hip_bf16.h>
#include <math.h>

// ---------------- constants from the reference ----------------
#define NPTS   65536
#define NIND   256
#define DIN    512
#define DEMB   512
#define NH     8
#define VD     64
#define EPSV   1e-5f
#define TMAXV  (10.0f / 65536.0f)          // 10 / 256^2
#define TMINV  (-TMAXV)
#define MAXTH  4.6051701859880914f         // -log(0.01)
#define MINTH  1.6094379124341003f         // -log(0.2)

typedef __bf16 bf16;
typedef bf16  v16bf  __attribute__((ext_vector_type(16)));
typedef bf16  bf16x8 __attribute__((ext_vector_type(8)));
typedef float v8f    __attribute__((ext_vector_type(8)));
typedef float f32x4  __attribute__((ext_vector_type(4)));

union Frag { v16bf v; bf16x8 h[2]; };

// ============================================================
// K0: theta_p / var_scale  [512]
// ============================================================
__global__ void k_params(const float* __restrict__ log_sigma,
                         const float* __restrict__ param_theta,
                         float* __restrict__ theta_p,
                         float* __restrict__ var_scale) {
    for (int c = threadIdx.x; c < DEMB; c += 256) {
        float pt  = param_theta[c];
        float sig = 1.0f / (1.0f + __expf(-pt));
        float th  = MINTH + sig * (MAXTH - MINTH);
        theta_p[c]   = th;
        var_scale[c] = 0.5f * __expf(2.0f * log_sigma[c]) / th;
    }
}

// ============================================================
// K1: WT[n][k] = bf16(weight[k][n]); WSQ[n][k] = bf16(w*w)
// ============================================================
__global__ void k_wcvt(const float* __restrict__ weight,
                       bf16* __restrict__ WT, bf16* __restrict__ WSQ) {
    int n = blockIdx.x;
    for (int k = threadIdx.x; k < DEMB; k += 256) {
        float w = weight[(size_t)k * DEMB + n];
        WT [(size_t)n * DEMB + k] = (bf16)w;
        WSQ[(size_t)n * DEMB + k] = (bf16)(w * w);
    }
}

// ============================================================
// K2: per inducing point m: t_q[m][h], mu_q[m][c]
// ============================================================
__global__ void k_zembed(const float* __restrict__ Z,
                         const float* __restrict__ W_embed,
                         const float* __restrict__ b_embed,
                         const float* __restrict__ W_mu,
                         const float* __restrict__ b_mu,
                         float* __restrict__ t_q,
                         float* __restrict__ mu_q) {
    __shared__ float Zs[DIN];
    int m = blockIdx.x;
    for (int k = threadIdx.x; k < DIN; k += 256)
        Zs[k] = Z[(size_t)m * DIN + k];
    __syncthreads();

    for (int c = threadIdx.x; c < DEMB; c += 256) {
        float acc = b_mu[c];
        for (int k = 0; k < DIN; ++k)
            acc += Zs[k] * W_mu[(size_t)k * DEMB + c];
        mu_q[(size_t)m * DEMB + c] = acc;
    }
    if (threadIdx.x < NH) {
        int h = threadIdx.x;
        float acc = b_embed[h];
        for (int k = 0; k < DIN; ++k)
            acc += Zs[k] * W_embed[(size_t)k * NH + h];
        acc = fminf(fmaxf(acc, TMINV), TMAXV);
        t_q[(size_t)m * NH + h] = acc;
    }
}

// ============================================================
// K3: per-head stable rank sort (M=256), write padded sorted_t
// ============================================================
__global__ void k_sort(const float* __restrict__ t_q,
                       float* __restrict__ sortedT,
                       int* __restrict__ order) {
    __shared__ float tv[NIND];
    int h = blockIdx.x;
    int i = threadIdx.x;
    tv[i] = t_q[(size_t)i * NH + h];
    __syncthreads();
    float ti = tv[i];
    int rank = 0;
    for (int j = 0; j < NIND; ++j) {
        float tj = tv[j];
        rank += (tj < ti || (tj == ti && j < i)) ? 1 : 0;
    }
    sortedT[(size_t)(rank + 1) * NH + h] = ti;
    order[(size_t)rank * NH + h] = i;
    if (i == 0) {
        sortedT[h] = TMINV - EPSV;
        sortedT[(size_t)(NIND + 1) * NH + h] = TMAXV + EPSV;
    }
}

// ============================================================
// K4: gathered + padded tables [258][512]: mu_s, var_q, cov_q
// ============================================================
__global__ void k_gather(const float* __restrict__ mu_q,
                         const float* __restrict__ lbd_q,
                         const float* __restrict__ tau_q,
                         const int* __restrict__ order,
                         float* __restrict__ muS,
                         float* __restrict__ varS,
                         float* __restrict__ covS) {
    int m = blockIdx.x;  // 0..257
    for (int c = threadIdx.x; c < DEMB; c += 256) {
        size_t o = (size_t)m * DEMB + c;
        if (m == 0 || m == NIND + 1) {
            muS[o] = 0.f; varS[o] = 0.f; covS[o] = 0.f;
        } else {
            int h = c >> 6;
            int j = order[(size_t)(m - 1) * NH + h];
            float mu = mu_q[(size_t)j * DEMB + c];
            float lb = lbd_q[(size_t)j * DEMB + c];
            float ta = tau_q[(size_t)j * DEMB + c];
            muS[o]  = mu;
            varS[o] = lb * lb + ta * ta;
            float cv = 0.f;
            if (m <= NIND - 1) {
                int j2 = order[(size_t)m * NH + h];
                cv = lb * tau_q[(size_t)j2 * DEMB + c];
            }
            covS[o] = cv;
        }
    }
}

// ============================================================
// K5: per-interval stats [257][512]: var_dt_all, diff_mu_all
// ============================================================
__global__ void k_intervals(const float* __restrict__ sortedT,
                            const float* __restrict__ muS,
                            const float* __restrict__ theta_p,
                            float* __restrict__ varDt,
                            float* __restrict__ diffMu) {
    int m = blockIdx.x;  // 0..256
    for (int c = threadIdx.x; c < DEMB; c += 256) {
        int h = c >> 6;
        float dt = sortedT[(size_t)(m + 1) * NH + h] - sortedT[(size_t)m * NH + h];
        float e  = __expf(-theta_p[c] * dt);
        size_t o = (size_t)m * DEMB + c;
        varDt[o]  = 1.0f - e * e;
        diffMu[o] = muS[(size_t)(m + 1) * DEMB + c] - e * muS[(size_t)m * DEMB + c];
    }
}

// ============================================================
// K6: per-point kernel: embed t, searchsorted, OU interpolation
//     32 points per block; writes bf16 A_mean / A_var [65536][512]
// ============================================================
__global__ void k_point(const float* __restrict__ X,
                        const float* __restrict__ W_embed,
                        const float* __restrict__ b_embed,
                        const float* __restrict__ sortedT,
                        const float* __restrict__ theta_p,
                        const float* __restrict__ var_scale,
                        const float* __restrict__ muS,
                        const float* __restrict__ varS,
                        const float* __restrict__ covS,
                        const float* __restrict__ varDt,
                        const float* __restrict__ diffMu,
                        bf16* __restrict__ Amean,
                        bf16* __restrict__ Avar) {
    __shared__ float WEs[DIN * NH];        // 16 KB
    __shared__ float sT[(NIND + 2) * NH];  // 8.25 KB
    __shared__ float bes[NH];
    __shared__ float Xs[32 * 128];         // 16 KB chunk
    __shared__ float sDtl[32 * NH];
    __shared__ float sDtr[32 * NH];
    __shared__ int   sIl[32 * NH];

    int tid = threadIdx.x;
    int n0 = blockIdx.x * 32;

    for (int i = tid; i < DIN * NH; i += 256) WEs[i] = W_embed[i];
    for (int i = tid; i < (NIND + 2) * NH; i += 256) sT[i] = sortedT[i];
    if (tid < NH) bes[tid] = b_embed[tid];
    __syncthreads();

    int p = tid >> 3;          // 0..31
    int h = tid & 7;           // 0..7
    float acc = 0.f;
    for (int kc = 0; kc < 4; ++kc) {
        for (int jj = 0; jj < 4; ++jj) {
            int idx = tid * 4 + jj;        // f32x4 index within chunk
            int p2  = idx >> 5;            // 32 f32x4 per row
            int k4  = idx & 31;
            // X is streamed exactly once: non-temporal load keeps L2 for tables
            ((f32x4*)Xs)[idx] = __builtin_nontemporal_load(
                (const f32x4*)(X + (size_t)(n0 + p2) * DIN + kc * 128) + k4);
        }
        __syncthreads();
        for (int k = 0; k < 128; ++k)
            acc += Xs[p * 128 + k] * WEs[(kc * 128 + k) * NH + h];
        __syncthreads();
    }
    float t = fminf(fmaxf(acc + bes[h], TMINV), TMAXV);

    // searchsorted(side='left') over sT[0..257] for this head
    int lo = 0, hi = NIND + 2;
    while (lo < hi) {
        int mid = (lo + hi) >> 1;
        if (sT[mid * NH + h] < t) lo = mid + 1; else hi = mid;
    }
    int ir = lo; if (ir < 1) ir = 1; if (ir > NIND + 1) ir = NIND + 1;
    int il = ir - 1;
    sDtl[tid] = t - sT[il * NH + h];
    sDtr[tid] = sT[ir * NH + h] - t;
    sIl[tid]  = il;
    __syncthreads();

    // phase 2: coalesced over columns
    for (int p2 = 0; p2 < 32; ++p2) {
        size_t n2 = (size_t)(n0 + p2);
        for (int c = tid; c < DEMB; c += 256) {
            int h2 = c >> 6;
            int sid = p2 * NH + h2;
            int iL = sIl[sid];
            int iR = iL + 1;
            float dtl = sDtl[sid], dtr = sDtr[sid];
            float th  = theta_p[c];
            float vsc = var_scale[c];
            float el = __expf(-th * dtl);
            float er = __expf(-th * dtr);
            float vdl = 1.f - el * el;
            float vdr = 1.f - er * er;
            size_t oL = (size_t)iL * DEMB + c;
            size_t oR = (size_t)iR * DEMB + c;
            float inv = 1.f / varDt[oL];
            float scale_r = vdl * inv;
            float gr = scale_r * er;
            float gl = vdr * el * inv;
            float mean = gr * diffMu[oL] + muS[oL] * el;
            float var  = gl * gl * varS[oL] + gr * gr * varS[oR]
                       + 2.f * gl * gr * covS[oL] + vsc * vdr * scale_r;
            // regular-temporal stores: Amean/Avar (128 MB total) should stay
            // resident in the 192 MB L2 for the GEMMs that follow
            Amean[n2 * DEMB + c] = (bf16)mean;
            Avar [n2 * DEMB + c] = (bf16)var;
        }
    }
}

// ============================================================
// K7: WMMA GEMM.  out[M,512] = A[M,512] @ B^T (B stored [N,K] bf16)
//     mode 0: + bias[n];  mode 1: max(·, EPS)
//     block = 256 threads = 8 waves; wave -> 64x64 tile (4 M x 4 N subtiles)
//     Single-buffered fragments (keeps static pressure ~200 VGPRs, under the
//     256 directly-addressable limit -> no scratch spills); the fully
//     unrolled K-loop lets SSA renaming hoist next-step loads over the
//     16-deep independent WMMA chain.
//     grid (N/64 = 8, M/512 = 128)
// ============================================================
__global__ void k_gemm(const bf16* __restrict__ A,
                       const bf16* __restrict__ B,
                       const float* __restrict__ bias,
                       float* __restrict__ out,
                       int mode) {
    int wave = threadIdx.x >> 5;
    int lane = threadIdx.x & 31;
    int half = lane >> 4;
    int lmod = lane & 15;
    int m0 = blockIdx.y * 512 + wave * 64;
    int n0 = blockIdx.x * 64;

    v8f c[4][4];
#pragma unroll
    for (int t = 0; t < 4; ++t)
#pragma unroll
        for (int s = 0; s < 4; ++s)
#pragma unroll
            for (int r = 0; r < 8; ++r) c[t][s][r] = 0.f;

    // per-lane base pointers (K-halved striping per ISA bf16 A/B layout)
    const bf16* Abase = A + (size_t)(m0 + lmod) * DEMB + half * 8;
    const bf16* Bbase = B + (size_t)(n0 + lmod) * DEMB + half * 8;

#pragma unroll
    for (int kk = 0; kk < DEMB / 32; ++kk) {
        int k0 = kk * 32;
        Frag a[4], b[4];
#pragma unroll
        for (int t = 0; t < 4; ++t) {
            const bf16* ap = Abase + (size_t)t * 16 * DEMB + k0;
            a[t].h[0] = *(const bf16x8*)(ap);
            a[t].h[1] = *(const bf16x8*)(ap + 16);
        }
#pragma unroll
        for (int s = 0; s < 4; ++s) {
            const bf16* bp = Bbase + (size_t)s * 16 * DEMB + k0;
            b[s].h[0] = *(const bf16x8*)(bp);
            b[s].h[1] = *(const bf16x8*)(bp + 16);
        }
        if (k0 + 128 < DEMB) {
            // pull the A stream ahead (global_prefetch_b8)
            __builtin_prefetch(Abase + k0 + 128, 0, 0);
        }
        // 16 independent WMMAs per K-step
#pragma unroll
        for (int s = 0; s < 4; ++s)
#pragma unroll
            for (int t = 0; t < 4; ++t)
                c[t][s] = __builtin_amdgcn_wmma_f32_16x16x32_bf16(
                    false, a[t].v, false, b[s].v, (short)0, c[t][s],
                    false, false);
    }

#pragma unroll
    for (int s = 0; s < 4; ++s) {
        int n = n0 + s * 16 + lmod;
        float bn = (mode == 0) ? bias[n] : 0.f;
#pragma unroll
        for (int t = 0; t < 4; ++t) {
#pragma unroll
            for (int r = 0; r < 8; ++r) {
                int m = m0 + t * 16 + half * 8 + r;
                float v = c[t][s][r];
                v = (mode == 0) ? (v + bn) : fmaxf(v, EPSV);
                // write-once output: non-temporal keeps A matrices hot in L2
                __builtin_nontemporal_store(v, out + (size_t)m * DEMB + n);
            }
        }
    }
}

// ============================================================
// launch
// ============================================================
extern "C" void kernel_launch(void* const* d_in, const int* in_sizes, int n_in,
                              void* d_out, int out_size, void* d_ws, size_t ws_size,
                              hipStream_t stream) {
    (void)in_sizes; (void)n_in; (void)out_size; (void)ws_size;
    const float* X        = (const float*)d_in[0];
    const float* W_embed  = (const float*)d_in[1];
    const float* b_embed  = (const float*)d_in[2];
    const float* Z        = (const float*)d_in[3];
    const float* W_mu     = (const float*)d_in[4];
    const float* b_mu     = (const float*)d_in[5];
    const float* lbd_q    = (const float*)d_in[6];
    const float* tau_q    = (const float*)d_in[7];
    const float* log_sig  = (const float*)d_in[8];
    const float* p_theta  = (const float*)d_in[9];
    const float* weight   = (const float*)d_in[10];
    const float* bias     = (const float*)d_in[11];
    float* out = (float*)d_out;

    // -------- workspace layout (~138 MB) --------
    char* w = (char*)d_ws;
    size_t o = 0;
    auto alloc = [&](size_t bytes) { char* p = w + o; o += (bytes + 255) & ~(size_t)255; return p; };
    bf16*  WT      = (bf16*) alloc((size_t)DEMB * DEMB * 2);
    bf16*  WSQ     = (bf16*) alloc((size_t)DEMB * DEMB * 2);
    float* theta_p = (float*)alloc(DEMB * 4);
    float* vscale  = (float*)alloc(DEMB * 4);
    float* t_q     = (float*)alloc((size_t)NIND * NH * 4);
    float* sortedT = (float*)alloc((size_t)(NIND + 2) * NH * 4);
    int*   order   = (int*)  alloc((size_t)NIND * NH * 4);
    float* mu_q    = (float*)alloc((size_t)NIND * DEMB * 4);
    float* muS     = (float*)alloc((size_t)(NIND + 2) * DEMB * 4);
    float* varS    = (float*)alloc((size_t)(NIND + 2) * DEMB * 4);
    float* covS    = (float*)alloc((size_t)(NIND + 2) * DEMB * 4);
    float* varDt   = (float*)alloc((size_t)(NIND + 1) * DEMB * 4);
    float* diffMu  = (float*)alloc((size_t)(NIND + 1) * DEMB * 4);
    bf16*  Amean   = (bf16*) alloc((size_t)NPTS * DEMB * 2);
    bf16*  Avar    = (bf16*) alloc((size_t)NPTS * DEMB * 2);

    k_params   <<<1,   256, 0, stream>>>(log_sig, p_theta, theta_p, vscale);
    k_wcvt     <<<DEMB, 256, 0, stream>>>(weight, WT, WSQ);
    k_zembed   <<<NIND, 256, 0, stream>>>(Z, W_embed, b_embed, W_mu, b_mu, t_q, mu_q);
    k_sort     <<<NH,  256, 0, stream>>>(t_q, sortedT, order);
    k_gather   <<<NIND + 2, 256, 0, stream>>>(mu_q, lbd_q, tau_q, order, muS, varS, covS);
    k_intervals<<<NIND + 1, 256, 0, stream>>>(sortedT, muS, theta_p, varDt, diffMu);
    k_point    <<<NPTS / 32, 256, 0, stream>>>(X, W_embed, b_embed, sortedT,
                                               theta_p, vscale, muS, varS, covS,
                                               varDt, diffMu, Amean, Avar);
    dim3 gg(DEMB / 64, NPTS / 512, 1);
    k_gemm<<<gg, 256, 0, stream>>>(Amean, WT,  bias, out,                       0);
    k_gemm<<<gg, 256, 0, stream>>>(Avar,  WSQ, bias, out + (size_t)NPTS * DEMB, 1);
}
